// FullAttention_48584670052733
// MI455X (gfx1250) — matrix operand order repaired and checked by reading, not compiled, not run on hardware
//
#include <hip/hip_runtime.h>
#include <hip/hip_bf16.h>

// ---------------------------------------------------------------------------
// MI455X / gfx1250, wave32.
//   prep:  f32 -> bf16 (Q, K natural; V transposed per head; FFN weights
//          pre-transposed into LDS-ready layouts)
//   k1:    dual-scale flash attention, double-buffered async LDS staging
//          (GLOBAL_LOAD_ASYNC_TO_LDS_B128 overlaps the WMMAs of the
//          previous tile; ASYNCcnt + split barriers)
//   k2:    src = FFN1(V) + FFN2(V1), phase-pipelined weight staging,
//          LayerNorm(64)
// All GEMMs: v_wmma_f32_16x16x32_bf16 (bf16 in, f32 accumulate).
// dp-projection == identity; attn_mask has no effect (see reference).
// ---------------------------------------------------------------------------

typedef __attribute__((ext_vector_type(16))) __bf16 v16bf;
typedef __attribute__((ext_vector_type(8)))  __bf16 v8bf;
typedef __attribute__((ext_vector_type(8)))  float  v8f;

#define BB 8
#define LL 1024
#define HH 8
#define EE 64
#define DFF 2048
#define NTOK (BB*LL*HH)
#define WB 8
#define TK 32
#define NKT (LL/TK)       // 32 key tiles

__device__ __forceinline__ v8f wmma_bf(v16bf a, v16bf b, v8f c) {
  return __builtin_amdgcn_wmma_f32_16x16x32_bf16(false, a, false, b,
                                                 (short)0, c, false, false);
}

// ---- CDNA5 async global->LDS copy (128b per lane), ASYNCcnt tracked ----
__device__ __forceinline__ unsigned lds_off32(const void* p) {
  return (unsigned)(size_t)p;   // LDS aperture: low 32 bits = LDS offset
}
__device__ __forceinline__ void async_g2l_b128(unsigned lds, const void* g) {
  asm volatile("global_load_async_to_lds_b128 %0, %1, off"
               :: "v"(lds), "v"(g) : "memory");
}
__device__ __forceinline__ void wait_async0() {
  asm volatile("s_wait_asynccnt 0" ::: "memory");
}

// ---- 16-lane-half xor butterflies via ds_swizzle (imm pattern) ----
__device__ __forceinline__ float xmax16(float x) {
  x = fmaxf(x, __int_as_float(__builtin_amdgcn_ds_swizzle(__float_as_int(x), 0x041f)));
  x = fmaxf(x, __int_as_float(__builtin_amdgcn_ds_swizzle(__float_as_int(x), 0x081f)));
  x = fmaxf(x, __int_as_float(__builtin_amdgcn_ds_swizzle(__float_as_int(x), 0x101f)));
  x = fmaxf(x, __int_as_float(__builtin_amdgcn_ds_swizzle(__float_as_int(x), 0x201f)));
  return x;
}
__device__ __forceinline__ float xsum16(float x) {
  x += __int_as_float(__builtin_amdgcn_ds_swizzle(__float_as_int(x), 0x041f));
  x += __int_as_float(__builtin_amdgcn_ds_swizzle(__float_as_int(x), 0x081f));
  x += __int_as_float(__builtin_amdgcn_ds_swizzle(__float_as_int(x), 0x101f));
  x += __int_as_float(__builtin_amdgcn_ds_swizzle(__float_as_int(x), 0x201f));
  return x;
}

// A-fragment (16x32): halfs j<8 -> K=8*hi+j ; j>=8 -> K=16+8*hi+(j-8)
__device__ __forceinline__ v16bf afrag16(const __bf16* base, int hi) {
  v8bf lo = *(const v8bf*)(base + 8*hi);
  v8bf hv = *(const v8bf*)(base + 16 + 8*hi);
  v16bf a;
#pragma unroll
  for (int j = 0; j < 8; ++j) { a[j] = lo[j]; a[j+8] = hv[j]; }
  return a;
}
// B-fragment (32x16): halfs j<8 -> K=8*hi+j ; j>=8 -> K=16+8*hi+(j-8)
__device__ __forceinline__ v16bf bfrag16(const __bf16* base, int hi) {
  v8bf lo = *(const v8bf*)(base + 8*hi);
  v8bf hv = *(const v8bf*)(base + 16 + 8*hi);
  v16bf a;
#pragma unroll
  for (int j = 0; j < 8; ++j) { a[j] = lo[j]; a[j+8] = hv[j]; }
  return a;
}

// ---------------------------------------------------------------------------
// prep kernels
// ---------------------------------------------------------------------------
__global__ __launch_bounds__(256, 1)
void conv_bf16_kernel(const float* __restrict__ src, __bf16* __restrict__ dst,
                      int n) {
  int i = (blockIdx.x * 256 + threadIdx.x) * 4;
  if (i < n) {
    float4 v = *(const float4*)(src + i);
    dst[i+0] = (__bf16)v.x; dst[i+1] = (__bf16)v.y;
    dst[i+2] = (__bf16)v.z; dst[i+3] = (__bf16)v.w;
  }
}

// Vt[b][h][e][l] = V[b][l][h][e]
__global__ __launch_bounds__(256, 1)
void conv_vt_kernel(const float* __restrict__ V, __bf16* __restrict__ Vt) {
  int i  = (blockIdx.x * 256 + threadIdx.x) * 4;
  int l  = i & 1023;
  int e  = (i >> 10) & 63;
  int hb = i >> 16;                                  // b*H + h
  const float* s = V + ((size_t)(hb >> 3) * LL + l) * (HH*EE)
                     + (size_t)(hb & 7) * EE + e;
#pragma unroll
  for (int j = 0; j < 4; ++j) Vt[i + j] = (__bf16)s[(size_t)j * HH * EE];
}

// W1t [dff][e] = W1^T ;  W2t [e][dff] = W2^T  (both FFNs)
__global__ __launch_bounds__(256, 1)
void conv_w_kernel(const float* __restrict__ W1a, const float* __restrict__ W1b,
                   const float* __restrict__ W2a, const float* __restrict__ W2b,
                   __bf16* __restrict__ W1ta, __bf16* __restrict__ W1tb,
                   __bf16* __restrict__ W2ta, __bf16* __restrict__ W2tb) {
  int i = blockIdx.x * 256 + threadIdx.x;
  int c = i >> 6, e = i & 63;
  W1ta[i] = (__bf16)W1a[(size_t)e * DFF + c];
  W1tb[i] = (__bf16)W1b[(size_t)e * DFF + c];
  int e2 = i >> 11, k = i & 2047;
  W2ta[i] = (__bf16)W2a[(size_t)k * EE + e2];
  W2tb[i] = (__bf16)W2b[(size_t)k * EE + e2];
}

// ---------------------------------------------------------------------------
// Kernel 1: dual-scale flash attention, double-buffered async staging.
// grid = B*H*(L/128) = 512 blocks, 256 threads.
// ---------------------------------------------------------------------------
__global__ __launch_bounds__(256, 1)
void attn_dual_kernel(const __bf16* __restrict__ Qb, const __bf16* __restrict__ Kb,
                      const __bf16* __restrict__ Vt,
                      __bf16* __restrict__ O0, __bf16* __restrict__ O1) {
  __shared__ __bf16 kbuf[2][TK * EE];     // [key][e], ping-pong
  __shared__ __bf16 vbuf[2][EE * TK];     // [e][key], ping-pong
  __shared__ __bf16 pbuf[WB][2][16 * TK];

  const int tid  = threadIdx.x;
  const int w    = tid >> 5;
  const int lane = tid & 31;
  const int m    = lane & 15;
  const int hi   = lane >> 4;

  const int qblk = blockIdx.x & 7;
  const int h    = (blockIdx.x >> 3) & 7;
  const int b    = blockIdx.x >> 6;
  const int l0   = qblk * 128 + w * 16;

  const float sc0 = 0.49f * 0.20412414523f; // (1-ema_w)^2 / sqrt(24)
  const float sc1 = 0.125f;                 // 1/sqrt(E)

  const __bf16* qrow = Qb + (((size_t)b * LL + (l0 + m)) * HH + h) * EE;
  const v16bf qa0 = afrag16(qrow,      hi);
  const v16bf qa1 = afrag16(qrow + 32, hi);

  v8f zz = {0.f, 0.f, 0.f, 0.f, 0.f, 0.f, 0.f, 0.f};
  v8f acc0[4], acc1[4];
  float mx0[8], sm0[8], mx1[8], sm1[8];
#pragma unroll
  for (int t = 0; t < 4; ++t) { acc0[t] = zz; acc1[t] = zz; }
#pragma unroll
  for (int r = 0; r < 8; ++r) { mx0[r] = mx1[r] = -1e30f; sm0[r] = sm1[r] = 0.f; }

  // per-lane async staging geometry
  const int krow = 4*w + (lane >> 3);     // 8 waves x 4 K rows = 32
  const int kch  = (lane & 7) * 8;
  const int ve   = 8*w + (lane >> 2);     // 8 waves x 8 e rows = 64
  const int vch  = (lane & 3) * 8;
  const __bf16* kgb = Kb + (((size_t)b*LL + krow)*HH + h)*EE + kch;
  const __bf16* vgb = Vt + ((size_t)(b*HH + h)*EE + ve)*LL + vch;
  const unsigned klds0 = lds_off32(kbuf[0] + krow*EE + kch);
  const unsigned klds1 = lds_off32(kbuf[1] + krow*EE + kch);
  const unsigned vlds0 = lds_off32(vbuf[0] + ve*TK  + vch);
  const unsigned vlds1 = lds_off32(vbuf[1] + ve*TK  + vch);

  // prologue: stage tile 0
  async_g2l_b128(klds0, kgb);
  async_g2l_b128(vlds0, vgb);

  for (int it = 0; it < NKT; ++it) {
    const int cur = it & 1;
    wait_async0();
    __syncthreads();            // tile `it` resident in buffer `cur`

    // overlap: stage tile it+1 into the other buffer under this tile's math
    if (it + 1 < NKT) {
      async_g2l_b128(cur ? klds0 : klds1, kgb + (size_t)(it+1)*TK*HH*EE);
      async_g2l_b128(cur ? vlds0 : vlds1, vgb + (it+1)*TK);
    }

    const __bf16* kb = kbuf[cur];
    const __bf16* vv = vbuf[cur];

    // ---- S = Q @ K^T ----
    v8f s0 = zz, s1 = zz;
    {
      const __bf16* kb0 = kb + m * EE;
      s0 = wmma_bf(qa0, bfrag16(kb0,      hi), s0);
      s0 = wmma_bf(qa1, bfrag16(kb0 + 32, hi), s0);
      const __bf16* kb1 = kb + (16 + m) * EE;
      s1 = wmma_bf(qa0, bfrag16(kb1,      hi), s1);
      s1 = wmma_bf(qa1, bfrag16(kb1 + 32, hi), s1);
    }

    float rmax[8];
#pragma unroll
    for (int r = 0; r < 8; ++r) rmax[r] = xmax16(fmaxf(s0[r], s1[r]));

    __bf16* pb0 = pbuf[w][0];
    __bf16* pb1 = pbuf[w][1];
    float f0[8], f1[8];
#pragma unroll
    for (int r = 0; r < 8; ++r) {
      float mn0 = fmaxf(mx0[r], rmax[r] * sc0);
      float mn1 = fmaxf(mx1[r], rmax[r] * sc1);
      f0[r] = __expf(mx0[r] - mn0); mx0[r] = mn0;
      f1[r] = __expf(mx1[r] - mn1); mx1[r] = mn1;
      float p00 = __expf(s0[r]*sc0 - mn0);
      float p01 = __expf(s1[r]*sc0 - mn0);
      float p10 = __expf(s0[r]*sc1 - mn1);
      float p11 = __expf(s1[r]*sc1 - mn1);
      int row = r + 8*hi;
      pb0[row*TK + m]      = (__bf16)p00;
      pb0[row*TK + 16 + m] = (__bf16)p01;
      pb1[row*TK + m]      = (__bf16)p10;
      pb1[row*TK + 16 + m] = (__bf16)p11;
      sm0[r] = sm0[r]*f0[r] + xsum16(p00 + p01);
      sm1[r] = sm1[r]*f1[r] + xsum16(p10 + p11);
    }
#pragma unroll
    for (int t = 0; t < 4; ++t)
#pragma unroll
      for (int r = 0; r < 8; ++r) { acc0[t][r] *= f0[r]; acc1[t][r] *= f1[r]; }

    // ---- acc += P @ V ----
    v16bf pa0 = afrag16(pb0 + m*TK, hi);
    v16bf pa1 = afrag16(pb1 + m*TK, hi);
#pragma unroll
    for (int t = 0; t < 4; ++t) {
      v16bf vb = bfrag16(vv + (16*t + m)*TK, hi);
      acc0[t] = wmma_bf(pa0, vb, acc0[t]);
      acc1[t] = wmma_bf(pa1, vb, acc1[t]);
    }
    __syncthreads();            // all reads of buffer `cur` done
  }

#pragma unroll
  for (int t = 0; t < 4; ++t) {
#pragma unroll
    for (int r = 0; r < 8; ++r) {
      int row = r + 8*hi;
      size_t o = (((size_t)b*LL + (l0 + row))*HH + h)*EE + 16*t + m;
      O0[o] = (__bf16)(acc0[t][r] / sm0[r]);
      O1[o] = (__bf16)(acc1[t][r] / sm1[r]);
    }
  }
}

// ---------------------------------------------------------------------------
// Kernel 2: src = FFN1(V) + FFN2(V1); LayerNorm(E=64).
// Phase-pipelined: FFN2(jb) weights load under FFN1(jb) compute and
// FFN1(jb+64) weights load under FFN2(jb) compute.
// ---------------------------------------------------------------------------
__global__ __launch_bounds__(256, 1)
void ffn_ln_kernel(const __bf16* __restrict__ X0, const __bf16* __restrict__ X1,
                   const __bf16* __restrict__ W1ta, const float* __restrict__ B1a,
                   const __bf16* __restrict__ W2ta, const float* __restrict__ B2a,
                   const __bf16* __restrict__ W1tb, const float* __restrict__ B1b,
                   const __bf16* __restrict__ W2tb, const float* __restrict__ B2b,
                   const float* __restrict__ LNW, const float* __restrict__ LNB,
                   float* __restrict__ out) {
  __shared__ __bf16 w1s0[64*64], w1s1[64*64];  // [dff_col][e]
  __shared__ __bf16 w2s0[64*64], w2s1[64*64];  // [e][dff_k]
  __shared__ __bf16 hb[WB][16 * 64];

  const int tid  = threadIdx.x;
  const int w    = tid >> 5;
  const int lane = tid & 31;
  const int m    = lane & 15;
  const int hi   = lane >> 4;
  const int t0   = blockIdx.x * 128 + w * 16;

  const __bf16* x0r = X0 + (size_t)(t0 + m) * EE;
  const __bf16* x1r = X1 + (size_t)(t0 + m) * EE;
  const v16bf x0a = afrag16(x0r,      hi);
  const v16bf x0b = afrag16(x0r + 32, hi);
  const v16bf x1a = afrag16(x1r,      hi);
  const v16bf x1b = afrag16(x1r + 32, hi);

  v8f zz = {0.f, 0.f, 0.f, 0.f, 0.f, 0.f, 0.f, 0.f};
  v8f acc[4];
#pragma unroll
  for (int t = 0; t < 4; ++t) acc[t] = zz;
  __bf16* hbw = hb[w];

  const int e2 = (lane >> 3);
  const int c2 = (lane & 7) * 8;

#define STAGE_F1(JB) do {                                                     \
    for (int j = 0; j < 2; ++j) {                                             \
      int t16 = w*2 + j;                                                      \
      int off = t16*256 + lane*8;                                             \
      async_g2l_b128(lds_off32(w1s0 + off), W1ta + (size_t)(JB)*EE + off);    \
      int e = t16*4 + e2;                                                     \
      async_g2l_b128(lds_off32(w2s0 + e*64 + c2),                             \
                     W2ta + (size_t)e*DFF + (JB) + c2);                       \
    } } while (0)
#define STAGE_F2(JB) do {                                                     \
    for (int j = 0; j < 2; ++j) {                                             \
      int t16 = w*2 + j;                                                      \
      int off = t16*256 + lane*8;                                             \
      async_g2l_b128(lds_off32(w1s1 + off), W1tb + (size_t)(JB)*EE + off);    \
      int e = t16*4 + e2;                                                     \
      async_g2l_b128(lds_off32(w2s1 + e*64 + c2),                             \
                     W2tb + (size_t)e*DFF + (JB) + c2);                       \
    } } while (0)

  STAGE_F1(0);   // prologue

  for (int jb = 0; jb < DFF; jb += 64) {
    wait_async0();
    __syncthreads();            // FFN1(jb) tiles resident
    STAGE_F2(jb);               // load FFN2(jb) under FFN1 compute

    // ---------------- FFN1 ----------------
#pragma unroll
    for (int t = 0; t < 4; ++t) {
      const __bf16* wb = w1s0 + (16*t + m) * 64;
      v8f hv = zz;
      hv = wmma_bf(x0a, bfrag16(wb,      hi), hv);
      hv = wmma_bf(x0b, bfrag16(wb + 32, hi), hv);
      float bias = B1a[jb + 16*t + m];
#pragma unroll
      for (int r = 0; r < 8; ++r) {
        float x = hv[r] + bias;
        float g = 0.5f * x * (1.f + erff(x * 0.70710678118f));
        hbw[(r + 8*hi)*64 + 16*t + m] = (__bf16)g;
      }
    }
    {
      v16bf ha0 = afrag16(hbw + m*64,      hi);
      v16bf ha1 = afrag16(hbw + m*64 + 32, hi);
#pragma unroll
      for (int t = 0; t < 4; ++t) {
        const __bf16* wc = w2s0 + (16*t + m) * 64;
        acc[t] = wmma_bf(ha0, bfrag16(wc,      hi), acc[t]);
        acc[t] = wmma_bf(ha1, bfrag16(wc + 32, hi), acc[t]);
      }
    }

    wait_async0();
    __syncthreads();            // FFN2(jb) tiles resident; FFN1 reads done
    if (jb + 64 < DFF) STAGE_F1(jb + 64);   // load next FFN1 under FFN2

    // ---------------- FFN2 ----------------
#pragma unroll
    for (int t = 0; t < 4; ++t) {
      const __bf16* wb = w1s1 + (16*t + m) * 64;
      v8f hv = zz;
      hv = wmma_bf(x1a, bfrag16(wb,      hi), hv);
      hv = wmma_bf(x1b, bfrag16(wb + 32, hi), hv);
      float bias = B1b[jb + 16*t + m];
#pragma unroll
      for (int r = 0; r < 8; ++r) {
        float x = hv[r] + bias;
        float g = 0.5f * x * (1.f + erff(x * 0.70710678118f));
        hbw[(r + 8*hi)*64 + 16*t + m] = (__bf16)g;
      }
    }
    {
      v16bf ha0 = afrag16(hbw + m*64,      hi);
      v16bf ha1 = afrag16(hbw + m*64 + 32, hi);
#pragma unroll
      for (int t = 0; t < 4; ++t) {
        const __bf16* wc = w2s1 + (16*t + m) * 64;
        acc[t] = wmma_bf(ha0, bfrag16(wc,      hi), acc[t]);
        acc[t] = wmma_bf(ha1, bfrag16(wc + 32, hi), acc[t]);
      }
    }
    __syncthreads();            // FFN2 reads done before its next restage
  }
#undef STAGE_F1
#undef STAGE_F2

  // ---- output biases, then LayerNorm over E=64 ----
#pragma unroll
  for (int t = 0; t < 4; ++t) {
    float bb = B2a[16*t + m] + B2b[16*t + m];
#pragma unroll
    for (int r = 0; r < 8; ++r) acc[t][r] += bb;
  }

  float mean[8], inv[8];
#pragma unroll
  for (int r = 0; r < 8; ++r) {
    float s = xsum16(acc[0][r] + acc[1][r] + acc[2][r] + acc[3][r]);
    float q = xsum16(acc[0][r]*acc[0][r] + acc[1][r]*acc[1][r]
                   + acc[2][r]*acc[2][r] + acc[3][r]*acc[3][r]);
    float mu  = s * (1.f/64.f);
    float var = q * (1.f/64.f) - mu*mu;
    mean[r] = mu;
    inv[r]  = rsqrtf(var + 1e-5f);
  }
#pragma unroll
  for (int t = 0; t < 4; ++t) {
    float wv = LNW[16*t + m], bv = LNB[16*t + m];
#pragma unroll
    for (int r = 0; r < 8; ++r) {
      int row = r + 8*hi;
      out[(size_t)(t0 + row)*EE + 16*t + m] =
          (acc[t][r] - mean[r]) * inv[r] * wv + bv;
    }
  }
}

// ---------------------------------------------------------------------------
extern "C" void kernel_launch(void* const* d_in, const int* in_sizes, int n_in,
                              void* d_out, int out_size, void* d_ws, size_t ws_size,
                              hipStream_t stream) {
  const float* Q   = (const float*)d_in[0];
  const float* K   = (const float*)d_in[1];
  const float* V   = (const float*)d_in[2];
  // d_in[3] attn_mask: no effect. d_in[4]/d_in[5] mlp_w/b: dp == identity.
  const float* W1a = (const float*)d_in[6];
  const float* B1a = (const float*)d_in[7];
  const float* W2a = (const float*)d_in[8];
  const float* B2a = (const float*)d_in[9];
  const float* W1b = (const float*)d_in[10];
  const float* B1b = (const float*)d_in[11];
  const float* W2b = (const float*)d_in[12];
  const float* B2b = (const float*)d_in[13];
  const float* LNW = (const float*)d_in[14];
  const float* LNB = (const float*)d_in[15];
  float* out = (float*)d_out;

  const size_t nqkv = (size_t)BB * LL * HH * EE;     // 4,194,304
  __bf16* X0   = (__bf16*)d_ws;
  __bf16* X1   = X0 + nqkv;
  __bf16* Qb   = X1 + nqkv;
  __bf16* Kb   = Qb + nqkv;
  __bf16* Vt   = Kb + nqkv;
  __bf16* W1ta = Vt + nqkv;
  __bf16* W1tb = W1ta + (size_t)DFF * EE;
  __bf16* W2ta = W1tb + (size_t)DFF * EE;
  __bf16* W2tb = W2ta + (size_t)DFF * EE;

  conv_bf16_kernel<<<4096, 256, 0, stream>>>(Q, Qb, (int)nqkv);
  conv_bf16_kernel<<<4096, 256, 0, stream>>>(K, Kb, (int)nqkv);
  conv_vt_kernel<<<4096, 256, 0, stream>>>(V, Vt);
  conv_w_kernel<<<512, 256, 0, stream>>>(W1a, W1b, W2a, W2b,
                                         W1ta, W1tb, W2ta, W2tb);

  attn_dual_kernel<<<512, 256, 0, stream>>>(Qb, Kb, Vt, X0, X1);
  ffn_ln_kernel<<<512, 256, 0, stream>>>(X0, X1, W1ta, B1a, W2ta, B2a,
                                         W1tb, B1b, W2tb, B2b, LNW, LNB, out);
}
